// ConvTokMWLA_85615878078713
// MI455X (gfx1250) — matrix operand order repaired and checked by reading, not compiled, not run on hardware
//
#include <hip/hip_runtime.h>
#include <math.h>

// ---------------- types ----------------
typedef __attribute__((ext_vector_type(16))) __bf16 v16bf;
typedef __attribute__((ext_vector_type(8)))  float  v8f;

#define WMMA_BF16(a, b, c) \
  __builtin_amdgcn_wmma_f32_16x16x32_bf16(false, (a), false, (b), (short)0, (c), false, false)

static __device__ __forceinline__ v16bf load_frag_pair(const __bf16* p0, const __bf16* p1) {
  union { v16bf v; uint4 q[2]; } u;
  u.q[0] = *reinterpret_cast<const uint4*>(p0);
  u.q[1] = *reinterpret_cast<const uint4*>(p1);
  return u.v;
}
static __device__ __forceinline__ v16bf load_frag_b(const __bf16* p) {
  union { v16bf v; uint4 q[2]; } u;
  u.q[0] = *reinterpret_cast<const uint4*>(p);
  u.q[1] = *reinterpret_cast<const uint4*>(p + 8);
  return u.v;
}

__constant__ int c_wsz[8] = {8, 16, 32, 64, 96, 128, 192, 256};

// ---------------- f32 -> bf16 convert ----------------
__global__ void k_convert(const float* __restrict__ s, __bf16* __restrict__ d, unsigned n) {
  unsigned i = blockIdx.x * 256u + threadIdx.x;
  if (i < n) d[i] = (__bf16)s[i];
}

// ---------------- tokenizer weight transpose: [447][32][16] -> [447][k*32+c] ----------------
__global__ void k_transpose_tokw(const float* __restrict__ w, float* __restrict__ wt) {
  unsigned i = blockIdx.x * 256u + threadIdx.x;
  if (i >= 447u * 512u) return;
  unsigned oc = i >> 9, kc = i & 511u, k = kc >> 5, c = kc & 31u;
  wt[i] = w[oc * 512u + c * 16u + k];
}

// ---------------- tokenize: conv(447) | mean(32) | std(32) | pos(1) ----------------
__global__ void k_tokenize(const float* __restrict__ x, const float* __restrict__ wt,
                           const float* __restrict__ tb, float* __restrict__ hout) {
  __shared__ float patch[512];
  int tok = blockIdx.x;            // b*512 + n
  int b = tok >> 9, n = tok & 511;
  int t = threadIdx.x;             // 128 threads
  const float* xp = x + (size_t)b * 262144 + (size_t)n * 512;  // (k,c) flat contiguous
  for (int i = t; i < 512; i += 128) patch[i] = xp[i];
  __syncthreads();
  float* hr = hout + (size_t)tok * 512;
  for (int oc = t; oc < 447; oc += 128) {
    float s = tb[oc];
    const float* wr = wt + (size_t)oc * 512;
    for (int i = 0; i < 512; ++i) s += patch[i] * wr[i];
    hr[oc] = s;
  }
  if (t < 32) {
    int c = t;
    float m = 0.f;
    for (int k = 0; k < 16; ++k) m += patch[k * 32 + c];
    m *= (1.f / 16.f);
    float q = 0.f;
    for (int k = 0; k < 16; ++k) { float d = patch[k * 32 + c] - m; q += d * d; }
    hr[447 + c] = m;
    hr[479 + c] = sqrtf(q * (1.f / 16.f));
  }
  if (t == 0) hr[511] = sinf(0.1f * (float)n);
}

// ---------------- generic bf16 WMMA GEMM: C = A[M,K] x W[N,K]^T + bias ----------------
// One wave (single-wave block, full register budget via launch_bounds) computes a
// 32(M) x 64(N) tile: 2 A-frags x 4 B-frags = 8 WMMAs per 32-wide k-step, register
// double-buffered so the next step's global_load_b128s are in flight while the
// current WMMAs execute. Requires K % 64 == 0 (holds: K = 512 or 2048).
// act: 0=none, 1=exact GELU
__global__ void __launch_bounds__(32, 1)
k_gemm(const __bf16* __restrict__ A, const __bf16* __restrict__ W,
       const float* __restrict__ bias, float* __restrict__ Cf,
       __bf16* __restrict__ Cb, int M, int N, int K, int act) {
  int lane = threadIdx.x;
  int lm = lane & 15, lh = lane >> 4;
  int n0 = blockIdx.x * 64;
  int m0 = blockIdx.y * 32;
  const __bf16* arow0 = A + (size_t)(m0 + lm) * K + lh * 8;        // M rows 0..15
  const __bf16* arow1 = A + (size_t)(m0 + 16 + lm) * K + lh * 8;   // M rows 16..31
  const __bf16* wrow[4];
#pragma unroll
  for (int j = 0; j < 4; ++j)
    wrow[j] = W + (size_t)(n0 + j * 16 + lm) * K + lh * 8;

  v8f c[2][4];
#pragma unroll
  for (int mi = 0; mi < 2; ++mi)
#pragma unroll
    for (int j = 0; j < 4; ++j) c[mi][j] = {};

  v16bf a0[2], a1[2], b0[4], b1[4];
  auto loadA = [&](int kk, v16bf* dst) {
    dst[0] = load_frag_pair(arow0 + kk, arow0 + kk + 16);
    dst[1] = load_frag_pair(arow1 + kk, arow1 + kk + 16);
  };
  auto loadB = [&](int kk, v16bf* dst) {
#pragma unroll
    for (int j = 0; j < 4; ++j) dst[j] = load_frag_pair(wrow[j] + kk, wrow[j] + kk + 16);
  };
  auto mma = [&](const v16bf* a, const v16bf* b) {
#pragma unroll
    for (int mi = 0; mi < 2; ++mi)
#pragma unroll
      for (int j = 0; j < 4; ++j) c[mi][j] = WMMA_BF16(a[mi], b[j], c[mi][j]);
  };

  loadA(0, a0);
  loadB(0, b0);
  for (int kk = 0; kk < K; kk += 64) {
    // stage kk+32 while computing kk
    loadA(kk + 32, a1);
    loadB(kk + 32, b1);
    __builtin_prefetch(arow0 + kk + 128, 0, 1);  // global_prefetch_b8, A stream ahead
    mma(a0, b0);
    if (kk + 64 < K) {  // stage kk+64 while computing kk+32
      loadA(kk + 64, a0);
      loadB(kk + 64, b0);
    }
    mma(a1, b1);
  }

#pragma unroll
  for (int mi = 0; mi < 2; ++mi) {
    int mrow = m0 + mi * 16 + lh * 8;
#pragma unroll
    for (int j = 0; j < 4; ++j) {
      int n = n0 + j * 16 + lm;
      float bv = bias ? bias[n] : 0.f;
#pragma unroll
      for (int r = 0; r < 8; ++r) {
        float v = c[mi][j][r] + bv;
        if (act == 1) v = 0.5f * v * (1.f + erff(v * 0.70710678118654752f));
        size_t idx = (size_t)(mrow + r) * N + n;
        if (Cf) Cf[idx] = v;
        else    Cb[idx] = (__bf16)v;
      }
    }
  }
}

// ---------------- V f32 [B*L,512] -> Vt bf16 [B][H][E=64][L=512] ----------------
__global__ void k_transpose_vt(const float* __restrict__ V, __bf16* __restrict__ Vt) {
  size_t i = (size_t)blockIdx.x * 256 + threadIdx.x;
  if (i >= (size_t)16 * 8 * 64 * 512) return;
  int s = (int)(i & 511); size_t t = i >> 9;
  int e = (int)(t & 63);  t >>= 6;
  int h = (int)(t & 7);   int b = (int)(t >> 3);
  Vt[i] = (__bf16)V[((size_t)(b * 512 + s)) * 512 + h * 64 + e];
}

// ---------------- banded flash attention: one wave per (b, head, 16-query tile) ----------------
__global__ void __launch_bounds__(32, 1)
k_attn(const __bf16* __restrict__ Q, const __bf16* __restrict__ Kb,
       const __bf16* __restrict__ Vt, float* __restrict__ Out) {
  __shared__ __bf16 pt[16 * 32];
  int lane = threadIdx.x, lm = lane & 15, lh = lane >> 4;
  int qt = blockIdx.x, h = blockIdx.y, b = blockIdx.z;
  int q0 = qt * 16;
  int W = c_wsz[h];
  const float scale = 0.125f;  // 1/sqrt(64)

  const __bf16* qrow = Q + ((size_t)(b * 512 + q0 + lm)) * 512 + h * 64;
  v16bf qa0 = load_frag_pair(qrow + 0  + lh * 8, qrow + 16 + lh * 8);
  v16bf qa1 = load_frag_pair(qrow + 32 + lh * 8, qrow + 48 + lh * 8);

  float rm[8], rs[8];
  v8f o[4];
  o[0] = {}; o[1] = {}; o[2] = {}; o[3] = {};
#pragma unroll
  for (int r = 0; r < 8; ++r) { rm[r] = -__builtin_inff(); rs[r] = 0.f; }

  int slo = q0 - W; if (slo < 0) slo = 0; slo &= ~31;
  int shi = q0 + 16 + W; if (shi > 512) shi = 512; shi = (shi + 31) & ~31;

  for (int ss = slo; ss < shi; ss += 32) {
    // prefetch next block's K rows
    if (ss + 32 < shi)
      __builtin_prefetch(Kb + ((size_t)(b * 512 + ss + 32 + lm)) * 512 + h * 64, 0, 1);
    // --- scores: S[16 x 32] = Q(16x64) x K^T ---
    v8f sc[2];
#pragma unroll
    for (int sj = 0; sj < 2; ++sj) {
      const __bf16* krow = Kb + ((size_t)(b * 512 + ss + sj * 16 + lm)) * 512 + h * 64;
      v16bf kb0 = load_frag_pair(krow + 0  + lh * 8, krow + 16 + lh * 8);
      v16bf kb1 = load_frag_pair(krow + 32 + lh * 8, krow + 48 + lh * 8);
      v8f s_ = {};
      s_ = WMMA_BF16(qa0, kb0, s_);
      s_ = WMMA_BF16(qa1, kb1, s_);
      sc[sj] = s_;
    }
    // --- online softmax over this 32-wide block ---
    float pv0[8], pv1[8];
#pragma unroll
    for (int r = 0; r < 8; ++r) {
      int m = q0 + lh * 8 + r;
      int n0_ = ss + lm, n1_ = ss + 16 + lm;
      bool v0ok = (abs(m - n0_) <= W);
      bool v1ok = (abs(m - n1_) <= W);
      float v0 = v0ok ? sc[0][r] * scale : -__builtin_inff();
      float v1 = v1ok ? sc[1][r] * scale : -__builtin_inff();
      float bm = fmaxf(v0, v1);
      for (int off = 1; off < 16; off <<= 1) bm = fmaxf(bm, __shfl_xor(bm, off, 32));
      float mnew = fmaxf(rm[r], bm);
      float corr = (mnew == -__builtin_inff()) ? 1.0f : expf(rm[r] - mnew);
      float p0 = v0ok ? expf(v0 - mnew) : 0.f;
      float p1 = v1ok ? expf(v1 - mnew) : 0.f;
      float bs = p0 + p1;
      for (int off = 1; off < 16; off <<= 1) bs += __shfl_xor(bs, off, 32);
      rs[r] = rs[r] * corr + bs;
      rm[r] = mnew;
      pv0[r] = p0; pv1[r] = p1;
#pragma unroll
      for (int j = 0; j < 4; ++j) o[j][r] *= corr;
    }
    // --- C-frag -> A-frag relayout of P through LDS ---
    __syncthreads();
#pragma unroll
    for (int r = 0; r < 8; ++r) {
      pt[(lh * 8 + r) * 32 + lm]      = (__bf16)pv0[r];
      pt[(lh * 8 + r) * 32 + 16 + lm] = (__bf16)pv1[r];
    }
    __syncthreads();
    v16bf pa = load_frag_b(&pt[lm * 32 + lh * 16]);
    // --- O += P(16x32) x V(32x64) ---
#pragma unroll
    for (int j = 0; j < 4; ++j) {
      const __bf16* vrow = Vt + ((size_t)((b * 8 + h) * 64 + j * 16 + lm)) * 512 + ss + lh * 16;
      v16bf vb = load_frag_b(vrow);
      o[j] = WMMA_BF16(pa, vb, o[j]);
    }
  }
#pragma unroll
  for (int j = 0; j < 4; ++j)
#pragma unroll
    for (int r = 0; r < 8; ++r) {
      int m = q0 + lh * 8 + r;
      int e = j * 16 + lm;
      Out[((size_t)(b * 512 + m)) * 512 + h * 64 + e] = o[j][r] / rs[r];
    }
}

// ---------------- residual + layernorm, one wave per row (D=512) ----------------
__global__ void k_addln(const float* __restrict__ X, const float* __restrict__ Y,
                        const float* __restrict__ g, const float* __restrict__ be,
                        float* __restrict__ Outp) {
  int row = blockIdx.x, lane = threadIdx.x;
  const float* xr = X + (size_t)row * 512;
  const float* yr = Y ? Y + (size_t)row * 512 : nullptr;
  float v[16];
  float s = 0.f;
#pragma unroll
  for (int j = 0; j < 16; ++j) {
    int i = lane + j * 32;
    v[j] = xr[i] + (yr ? yr[i] : 0.f);
    s += v[j];
  }
  for (int off = 1; off < 32; off <<= 1) s += __shfl_xor(s, off, 32);
  float mean = s * (1.f / 512.f);
  float q = 0.f;
#pragma unroll
  for (int j = 0; j < 16; ++j) { float d = v[j] - mean; q += d * d; }
  for (int off = 1; off < 32; off <<= 1) q += __shfl_xor(q, off, 32);
  float inv = rsqrtf(q * (1.f / 512.f) + 1e-5f);
  float* outr = Outp + (size_t)row * 512;
#pragma unroll
  for (int j = 0; j < 16; ++j) {
    int i = lane + j * 32;
    outr[i] = (v[j] - mean) * inv * g[i] + be[i];
  }
}

// ---------------- mean pool over L + classifier head ----------------
__global__ void k_poolhead(const float* __restrict__ enc, const float* __restrict__ hw,
                           const float* __restrict__ hbias, float* __restrict__ y) {
  __shared__ float pooled[512];
  int b = blockIdx.x, t = threadIdx.x;  // 256 threads
  for (int d = t; d < 512; d += 256) {
    float s = 0.f;
    const float* base = enc + ((size_t)b * 512) * 512 + d;
    for (int l = 0; l < 512; ++l) s += base[(size_t)l * 512];
    pooled[d] = s * (1.f / 512.f);
  }
  __syncthreads();
  if (t < 10) {
    float s = hbias[t];
    for (int d = 0; d < 512; ++d) s += pooled[d] * hw[t * 512 + d];
    y[b * 10 + t] = s;
  }
}

// ---------------- host orchestration ----------------
extern "C" void kernel_launch(void* const* d_in, const int* in_sizes, int n_in,
                              void* d_out, int out_size, void* d_ws, size_t ws_size,
                              hipStream_t stream) {
  (void)in_sizes; (void)n_in; (void)out_size; (void)ws_size;
  const float* x    = (const float*)d_in[0];
  const float* tokw = (const float*)d_in[1];
  const float* tokb = (const float*)d_in[2];
  const float* wq = (const float*)d_in[3];  const float* bq = (const float*)d_in[4];
  const float* wk = (const float*)d_in[5];  const float* bk = (const float*)d_in[6];
  const float* wv = (const float*)d_in[7];  const float* bv = (const float*)d_in[8];
  const float* wo = (const float*)d_in[9];  const float* bo = (const float*)d_in[10];
  const float* w1 = (const float*)d_in[11]; const float* b1 = (const float*)d_in[12];
  const float* w2 = (const float*)d_in[13]; const float* b2 = (const float*)d_in[14];
  const float* ln1g = (const float*)d_in[15]; const float* ln1b = (const float*)d_in[16];
  const float* ln2g = (const float*)d_in[17]; const float* ln2b = (const float*)d_in[18];
  const float* lnfg = (const float*)d_in[19]; const float* lnfb = (const float*)d_in[20];
  const float* hw = (const float*)d_in[21];   const float* hbb = (const float*)d_in[22];

  const int M = 16 * 512;      // 8192 tokens
  const size_t DD = 512 * 512; // 262144

  unsigned char* base = (unsigned char*)d_ws;
  size_t off = 0;
  auto alloc = [&](size_t bytes) { size_t o = off; off = (off + bytes + 255) & ~(size_t)255; return o; };

  __bf16* wq_bf = (__bf16*)(base + alloc(3 * DD * 2));
  __bf16* wk_bf = (__bf16*)(base + alloc(3 * DD * 2));
  __bf16* wv_bf = (__bf16*)(base + alloc(3 * DD * 2));
  __bf16* wo_bf = (__bf16*)(base + alloc(3 * DD * 2));
  __bf16* w1_bf = (__bf16*)(base + alloc((size_t)3 * 2048 * 512 * 2));
  __bf16* w2_bf = (__bf16*)(base + alloc((size_t)3 * 512 * 2048 * 2));
  float*  tokwT = (float*)(base + alloc((size_t)447 * 512 * 4));
  float*  h     = (float*)(base + alloc((size_t)M * 512 * 4));
  float*  tmp   = (float*)(base + alloc((size_t)M * 512 * 4));
  float*  attn  = (float*)(base + alloc((size_t)M * 512 * 4));
  __bf16* hb    = (__bf16*)(base + alloc((size_t)M * 512 * 2));
  __bf16* qb    = (__bf16*)(base + alloc((size_t)M * 512 * 2));
  __bf16* kb    = (__bf16*)(base + alloc((size_t)M * 512 * 2));
  __bf16* vt    = (__bf16*)(base + alloc((size_t)M * 512 * 2));
  __bf16* f1b   = (__bf16*)(base + alloc((size_t)M * 2048 * 2));

  auto cvt = [&](const float* s, __bf16* d, size_t n) {
    k_convert<<<dim3((unsigned)((n + 255) / 256)), dim3(256), 0, stream>>>(s, d, (unsigned)n);
  };

  // weight conversions (once per launch; deterministic)
  cvt(wq, wq_bf, 3 * DD); cvt(wk, wk_bf, 3 * DD);
  cvt(wv, wv_bf, 3 * DD); cvt(wo, wo_bf, 3 * DD);
  cvt(w1, w1_bf, (size_t)3 * 2048 * 512);
  cvt(w2, w2_bf, (size_t)3 * 512 * 2048);
  k_transpose_tokw<<<dim3((447 * 512 + 255) / 256), dim3(256), 0, stream>>>(tokw, tokwT);

  // tokenize -> h
  k_tokenize<<<dim3(M), dim3(128), 0, stream>>>(x, tokwT, tokb, h);

  dim3 gProj(512 / 64, M / 32);   // N=512 GEMMs, 32-row M tiles
  dim3 gFf1(2048 / 64, M / 32);   // N=2048
  dim3 blk(32);

  for (int l = 0; l < 3; ++l) {
    const __bf16* wql = wq_bf + (size_t)l * DD;
    const __bf16* wkl = wk_bf + (size_t)l * DD;
    const __bf16* wvl = wv_bf + (size_t)l * DD;
    const __bf16* wol = wo_bf + (size_t)l * DD;
    const __bf16* w1l = w1_bf + (size_t)l * 2048 * 512;
    const __bf16* w2l = w2_bf + (size_t)l * 512 * 2048;

    cvt(h, hb, (size_t)M * 512);
    k_gemm<<<gProj, blk, 0, stream>>>(hb, wql, bq + l * 512, nullptr, qb, M, 512, 512, 0);
    k_gemm<<<gProj, blk, 0, stream>>>(hb, wkl, bk + l * 512, nullptr, kb, M, 512, 512, 0);
    k_gemm<<<gProj, blk, 0, stream>>>(hb, wvl, bv + l * 512, tmp, nullptr, M, 512, 512, 0);
    k_transpose_vt<<<dim3((unsigned)(((size_t)M * 512 + 255) / 256)), dim3(256), 0, stream>>>(tmp, vt);
    k_attn<<<dim3(32, 8, 16), blk, 0, stream>>>(qb, kb, vt, attn);
    cvt(attn, qb, (size_t)M * 512);  // reuse qb as bf16 attention output
    k_gemm<<<gProj, blk, 0, stream>>>(qb, wol, bo + l * 512, tmp, nullptr, M, 512, 512, 0);
    k_addln<<<dim3(M), blk, 0, stream>>>(h, tmp, ln1g + l * 512, ln1b + l * 512, h);
    cvt(h, hb, (size_t)M * 512);
    k_gemm<<<gFf1, blk, 0, stream>>>(hb, w1l, b1 + l * 2048, nullptr, f1b, M, 2048, 512, 1);
    k_gemm<<<gProj, blk, 0, stream>>>(f1b, w2l, b2 + l * 512, tmp, nullptr, M, 512, 2048, 0);
    k_addln<<<dim3(M), blk, 0, stream>>>(h, tmp, ln2g + l * 512, ln2b + l * 512, h);
  }

  // final LN -> enc region of d_out, then pooled head -> y region
  float* out = (float*)d_out;
  float* enc = out + 160;  // (y[16,10], enc[16,512,512]) concatenated
  k_addln<<<dim3(M), blk, 0, stream>>>(h, nullptr, lnfg, lnfb, enc);
  k_poolhead<<<dim3(16), dim3(256), 0, stream>>>(enc, hw, hbb, out);
}
